// NMT_61272003444928
// MI455X (gfx1250) — matrix operand order repaired and checked by reading, not compiled, not run on hardware
//
#include <hip/hip_runtime.h>

// ---------------- model dims ----------------
#define TSEQ   50
#define NB     64
#define EMB    300
#define EMB_P  320          // padded K for embedding GEMMs
#define ENC_H  512
#define DEC_H  1024
#define TGT_V  23262
#define INP_KP 1344         // 1324 (ct||pe) padded to /32
#define ACT_K  2368         // 1344 + 1024 (combined [inp, h] row)
#define GEN_NP 23296        // 23262 padded to /128

#define TM 64
#define TN 128

typedef __attribute__((ext_vector_type(16))) __bf16 v16bf;
typedef __attribute__((ext_vector_type(8)))  float  v8f;

__device__ __forceinline__ unsigned short f2bf(float f) {
  unsigned int u = __float_as_uint(f);
  u += 0x7FFFu + ((u >> 16) & 1u);           // round-to-nearest-even
  return (unsigned short)(u >> 16);
}
__device__ __forceinline__ float sigf(float x) { return 1.0f / (1.0f + expf(-x)); }

// CDNA5 async global->LDS DMA (ASYNCcnt-tracked). lds_off = LDS byte offset
// (low 32 bits of the generic shared-memory address), gaddr = 64-bit global.
__device__ __forceinline__ void async_b128(unsigned lds_off, unsigned long long gaddr) {
  asm volatile("global_load_async_to_lds_b128 %0, %1, off"
               :: "v"(lds_off), "v"(gaddr) : "memory");
}
__device__ __forceinline__ unsigned lds_off32(const void* p) {
  return (unsigned)(size_t)p;                // flat LDS aperture: low word = LDS offset
}

// ======================================================================
// Generic bf16 WMMA GEMM:  out(MxN) = A(MxK) * B(NxK)^T [+bias0+bias1]
//                                      [+addsrc] [tanh] -> f32 / bf16
// Block: 64x128 tile, 256 threads = 8 wave32 waves; each wave owns a
// 16x64 strip = 4 of the 32 16x16 WMMA tiles. K staged in 32-deep chunks
// through double-buffered LDS via GLOBAL_LOAD_ASYNC_TO_LDS_B128.
// M mult of 64, N mult of 128 (grid), K mult of 32.
// ======================================================================
__global__ void __launch_bounds__(256)
nmt_gemm_bf16(const unsigned short* __restrict__ A, int lda,
              const unsigned short* __restrict__ B, int ldb,
              const float* __restrict__ bias0, const float* __restrict__ bias1, int nbias,
              const float* __restrict__ addsrc, int ldadd,
              float* __restrict__ outf, unsigned short* __restrict__ outb, int ldo,
              int K, int act)
{
  __shared__ __align__(16) unsigned short As[2][TM * 32];
  __shared__ __align__(16) unsigned short Bs[2][TN * 32];
  const int m0   = blockIdx.y * TM;
  const int n0   = blockIdx.x * TN;
  const int tid  = threadIdx.x;
  const int w    = tid >> 5, lane = tid & 31;
  const int mt   = w & 3;          // M-tile 0..3
  const int ntp  = w >> 2;         // N-tile quad 0..1 (owns 4 consecutive N tiles)
  const int half = lane >> 4, lr16 = lane & 15;

  // staging assignment: 16B per transfer; A chunk = 4KB, B chunk = 8KB
  const int srow = tid >> 2;             // 0..63
  const int scol = (tid & 3) << 3;       // 0,8,16,24 (ushorts)
  const unsigned aoff  = lds_off32(&As[0][srow * 32 + scol]);
  const unsigned boff1 = lds_off32(&Bs[0][srow * 32 + scol]);
  const unsigned boff2 = boff1 + 64 * 32 * 2;
  const unsigned long long gA  = (unsigned long long)(size_t)(A + (size_t)(m0 + srow) * lda + scol);
  const unsigned long long gB1 = (unsigned long long)(size_t)(B + (size_t)(n0 + srow) * ldb + scol);
  const unsigned long long gB2 = (unsigned long long)(size_t)(B + (size_t)(n0 + 64 + srow) * ldb + scol);

  union { v8f v; float f[8]; } acc[4];
#pragma unroll
  for (int t = 0; t < 4; ++t)
#pragma unroll
    for (int i = 0; i < 8; ++i) acc[t].f[i] = 0.f;

  const unsigned int* As32 = (const unsigned int*)&As[0][0];
  const unsigned int* Bs32 = (const unsigned int*)&Bs[0][0];
  const int nk = K >> 5;

  // prologue: stage chunk 0 into buffer 0
  async_b128(aoff, gA);
  async_b128(boff1, gB1);
  async_b128(boff2, gB2);

  for (int ik = 0; ik < nk; ++ik) {
    const int buf = ik & 1;
    __syncthreads();                              // readers of buf^1 are done
    if (ik + 1 < nk) {
      const unsigned long long ko = (unsigned long long)((ik + 1) << 5) * 2ull; // bytes
      const unsigned bo = (unsigned)(buf ^ 1);
      async_b128(aoff + bo * (TM * 32 * 2), gA + ko);
      async_b128(boff1 + bo * (TN * 32 * 2), gB1 + ko);
      async_b128(boff2 + bo * (TN * 32 * 2), gB2 + ko);
      asm volatile("s_wait_asynccnt 3" ::: "memory");   // chunk ik has landed
    } else {
      asm volatile("s_wait_asynccnt 0" ::: "memory");
    }
    __syncthreads();                              // everyone's chunk ik is visible

    // A frag (16x32 bf16): lane<16 -> K 0..15; lane>=16 -> K 16..31
    union { v16bf v; unsigned int u[8]; } a;
    const int abase = buf * (TM * 32 / 2) + (mt * 16 + lr16) * 16;  // uints
#pragma unroll
    for (int j = 0; j < 4; ++j) {
      a.u[j]     = As32[abase + 8 * half + j];
      a.u[4 + j] = As32[abase + 8 * half + 4 + j];
    }
    // Batch all 4 B fragments into distinct registers first so the ds_loads
    // can be issued back-to-back and overlapped with the WMMA drain.
    union { v16bf v; unsigned int u[8]; } bfr[4];
    const int bbufu = buf * (TN * 32 / 2);
#pragma unroll
    for (int t = 0; t < 4; ++t) {
      // B frag (32x16): lane = N col; lanes<16 hold K0..15, lanes>=16 hold K16..31
      const int nb = bbufu + ((ntp * 4 + t) * 16 + lr16) * 16 + 8 * half;
#pragma unroll
      for (int j = 0; j < 8; ++j) bfr[t].u[j] = Bs32[nb + j];
    }
#pragma unroll
    for (int t = 0; t < 4; ++t) {
      acc[t].v = __builtin_amdgcn_wmma_f32_16x16x32_bf16(false, a.v, false, bfr[t].v,
                                                         (short)0, acc[t].v, false, false);
    }
  }

#pragma unroll
  for (int t = 0; t < 4; ++t) {
    const int n = n0 + (ntp * 4 + t) * 16 + lr16;
    float bsum = 0.f;
    if (bias0 && n < nbias) bsum += bias0[n];
    if (bias1 && n < nbias) bsum += bias1[n];
#pragma unroll
    for (int r = 0; r < 8; ++r) {
      const int m = m0 + mt * 16 + half * 8 + r;   // C/D layout: VGPR r -> M = r + 8*half
      float v = acc[t].f[r] + bsum;
      if (addsrc) v += addsrc[(size_t)m * ldadd + n];
      if (act == 1) v = tanhf(v);
      if (outf) outf[(size_t)m * ldo + n] = v;
      if (outb) outb[(size_t)m * ldo + n] = f2bf(v);
    }
  }
}

// ---------------- helpers ----------------
__global__ void nmt_convert(const float* __restrict__ src, int R, int C,
                            unsigned short* __restrict__ dst, int ldd, int col0,
                            int Rp, int Cp)
{
  long i = (long)blockIdx.x * 256 + threadIdx.x;
  if (i >= (long)Rp * Cp) return;
  int r = (int)(i / Cp), c = (int)(i % Cp);
  float v = (r < R && c < C) ? src[(long)r * C + c] : 0.f;
  dst[(long)r * ldd + col0 + c] = f2bf(v);
}

__global__ void nmt_embed_src(const int* __restrict__ src, const float* __restrict__ emb,
                              unsigned short* __restrict__ xbf)
{
  long i = (long)blockIdx.x * 256 + threadIdx.x;           // over 3200*320
  if (i >= (long)TSEQ * NB * EMB_P) return;
  int row = (int)(i / EMB_P), e = (int)(i % EMB_P);
  float v = (e < EMB) ? emb[(long)src[row] * EMB + e] : 0.f;
  xbf[i] = f2bf(v);
}

__global__ void nmt_zero(float* p, long n) {
  long i = (long)blockIdx.x * 256 + threadIdx.x;
  if (i < n) p[i] = 0.f;
}

__global__ void nmt_lstm_cell(const float* __restrict__ gates, int H,
                              float* __restrict__ h, float* __restrict__ c,
                              unsigned short* b0, int ld0,
                              unsigned short* b1, int ld1,
                              unsigned short* b2, int ld2,
                              float* __restrict__ hsout)
{
  int i = blockIdx.x * 256 + threadIdx.x;
  if (i >= NB * H) return;
  int b = i / H, j = i % H;
  const float* g = gates + (long)b * 4 * H;
  float gi = g[j], gf = g[H + j], gg = g[2 * H + j], go = g[3 * H + j];
  float cn = sigf(gf) * c[i] + sigf(gi) * tanhf(gg);
  float hn = sigf(go) * tanhf(cn);
  c[i] = cn; h[i] = hn;
  unsigned short hb = f2bf(hn);
  if (b0) b0[(long)b * ld0 + j] = hb;
  if (b1) b1[(long)b * ld1 + j] = hb;
  if (b2) b2[(long)b * ld2 + j] = hb;
  if (hsout) hsout[i] = hn;
}

__global__ void nmt_encout(const float* __restrict__ hsF, const float* __restrict__ hsR,
                           float* __restrict__ eo)
{
  long i = (long)blockIdx.x * 256 + threadIdx.x;           // 50*64*512
  if (i >= (long)TSEQ * NB * ENC_H) return;
  int k = (int)(i % ENC_H);
  long tb = i / ENC_H;                                     // t*64+b
  int t = (int)(tb / NB), b = (int)(tb % NB);
  eo[tb * DEC_H + k]          = hsF[i];
  eo[tb * DEC_H + ENC_H + k]  = hsR[((long)(TSEQ - 1 - t) * NB + b) * ENC_H + k];
}

__global__ void nmt_h0(const float* hF, const float* cF, const float* hR, const float* cR,
                       float* hdec, float* cdec,
                       unsigned short* hq, unsigned short* actb, unsigned short* catb)
{
  int i = blockIdx.x * 256 + threadIdx.x;                  // 64*512
  if (i >= NB * ENC_H) return;
  int b = i / ENC_H, k = i % ENC_H;
  int j0 = 2 * k, j1 = 2 * k + 1;
  long base = (long)b * DEC_H;
  hdec[base + j0] = hF[i]; hdec[base + j1] = hR[i];
  cdec[base + j0] = cF[i]; cdec[base + j1] = cR[i];
  unsigned short h0b = f2bf(hF[i]), h1b = f2bf(hR[i]);
  hq[base + j0] = h0b;                         hq[base + j1] = h1b;
  actb[(long)b * ACT_K + INP_KP + j0] = h0b;   actb[(long)b * ACT_K + INP_KP + j1] = h1b;
  catb[(long)b * 2048 + DEC_H + j0]   = h0b;   catb[(long)b * 2048 + DEC_H + j1]   = h1b;
}

// attention: score = <enc_out[t,b,:], q[b,:]>, softmax over t, context st -> cat[:,0:1024] (bf16)
// also builds pe = dec_emb[trg[b]] into act[:,1024:1344]
__global__ void __launch_bounds__(256)
nmt_attn(const float* __restrict__ q, const float* __restrict__ eo,
         const float* __restrict__ dec_emb, const int* __restrict__ trg_row,
         unsigned short* __restrict__ catb, unsigned short* __restrict__ actb)
{
  int b = blockIdx.x, tid = threadIdx.x;
  int w = tid >> 5, lane = tid & 31;
  __shared__ float sc[64];
  const float* qb = q + (long)b * DEC_H;
  for (int t = w; t < TSEQ; t += 8) {
    const float* er = eo + ((long)t * NB + b) * DEC_H;
    float p = 0.f;
    for (int h = lane; h < DEC_H; h += 32) p += er[h] * qb[h];
    for (int off = 16; off; off >>= 1) p += __shfl_xor(p, off);
    if (lane == 0) sc[t] = p;
  }
  __syncthreads();
  if (tid == 0) {
    float mx = sc[0];
    for (int t = 1; t < TSEQ; ++t) mx = fmaxf(mx, sc[t]);
    float s = 0.f;
    for (int t = 0; t < TSEQ; ++t) { float e = expf(sc[t] - mx); sc[t] = e; s += e; }
    float inv = 1.f / s;
    for (int t = 0; t < TSEQ; ++t) sc[t] *= inv;
  }
  __syncthreads();
  for (int h = tid; h < DEC_H; h += 256) {
    float acc = 0.f;
    for (int t = 0; t < TSEQ; ++t) acc += sc[t] * eo[((long)t * NB + b) * DEC_H + h];
    catb[(long)b * 2048 + h] = f2bf(acc);
  }
  int tok = trg_row[b];
  for (int e = tid; e < EMB_P; e += 256) {
    float v = (e < EMB) ? dec_emb[(long)tok * EMB + e] : 0.f;
    actb[(long)b * ACT_K + DEC_H + e] = f2bf(v);
  }
}

__global__ void __launch_bounds__(256)
nmt_logsoftmax(const float* __restrict__ logits, int ldl, int Nv,
               float* __restrict__ outp, int ldo)
{
  int b = blockIdx.x, tid = threadIdx.x;
  __shared__ float red[256];
  __shared__ float s_max, s_lse;
  const float* row = logits + (long)b * ldl;
  float m = -1e30f;
  for (int n = tid; n < Nv; n += 256) m = fmaxf(m, row[n]);
  red[tid] = m; __syncthreads();
  for (int s = 128; s; s >>= 1) { if (tid < s) red[tid] = fmaxf(red[tid], red[tid + s]); __syncthreads(); }
  if (!tid) s_max = red[0];
  __syncthreads();
  float sum = 0.f;
  for (int n = tid; n < Nv; n += 256) sum += expf(row[n] - s_max);
  red[tid] = sum; __syncthreads();
  for (int s = 128; s; s >>= 1) { if (tid < s) red[tid] += red[tid + s]; __syncthreads(); }
  if (!tid) s_lse = logf(red[0]);
  __syncthreads();
  float* orow = outp + (long)b * ldo;
  for (int n = tid; n < Nv; n += 256) orow[n] = row[n] - s_max - s_lse;
}

// ======================================================================
extern "C" void kernel_launch(void* const* d_in, const int* in_sizes, int n_in,
                              void* d_out, int out_size, void* d_ws, size_t ws_size,
                              hipStream_t stream)
{
  (void)in_sizes; (void)n_in; (void)out_size; (void)ws_size;
  const int*   src     = (const int*)  d_in[0];
  const int*   trg     = (const int*)  d_in[1];
  const float* eWihF   = (const float*)d_in[2];
  const float* eWhhF   = (const float*)d_in[3];
  const float* ebihF   = (const float*)d_in[4];
  const float* ebhhF   = (const float*)d_in[5];
  const float* eWihR   = (const float*)d_in[6];
  const float* eWhhR   = (const float*)d_in[7];
  const float* ebihR   = (const float*)d_in[8];
  const float* ebhhR   = (const float*)d_in[9];
  const float* dWih    = (const float*)d_in[10];
  const float* dWhh    = (const float*)d_in[11];
  const float* dbih    = (const float*)d_in[12];
  const float* dbhh    = (const float*)d_in[13];
  const float* enc_emb = (const float*)d_in[14];
  const float* dec_emb = (const float*)d_in[15];
  const float* wi      = (const float*)d_in[16];
  const float* wo      = (const float*)d_in[17];
  const float* gen_w   = (const float*)d_in[18];
  const float* gen_b   = (const float*)d_in[19];
  float* out = (float*)d_out;

  char* base = (char*)d_ws; size_t off = 0;
  auto alloc = [&](size_t bytes) -> char* {
    char* p = base + off; off = (off + bytes + 255) & ~(size_t)255; return p;
  };
  unsigned short* xbf    = (unsigned short*)alloc((size_t)TSEQ * NB * EMB_P * 2);
  unsigned short* WihFb  = (unsigned short*)alloc((size_t)2048 * EMB_P * 2);
  unsigned short* WihRb  = (unsigned short*)alloc((size_t)2048 * EMB_P * 2);
  unsigned short* WhhFb  = (unsigned short*)alloc((size_t)2048 * ENC_H * 2);
  unsigned short* WhhRb  = (unsigned short*)alloc((size_t)2048 * ENC_H * 2);
  unsigned short* Wcomb  = (unsigned short*)alloc((size_t)4096 * ACT_K * 2);
  unsigned short* wib    = (unsigned short*)alloc((size_t)1024 * 1024 * 2);
  unsigned short* wob    = (unsigned short*)alloc((size_t)1024 * 2048 * 2);
  unsigned short* genwb  = (unsigned short*)alloc((size_t)GEN_NP * 1024 * 2);
  float* xpF    = (float*)alloc((size_t)TSEQ * NB * 2048 * 4);
  float* xpR    = (float*)alloc((size_t)TSEQ * NB * 2048 * 4);
  float* hsF    = (float*)alloc((size_t)TSEQ * NB * ENC_H * 4);
  float* hsR    = (float*)alloc((size_t)TSEQ * NB * ENC_H * 4);
  float* eo     = (float*)alloc((size_t)TSEQ * NB * DEC_H * 4);
  float* hF     = (float*)alloc((size_t)NB * ENC_H * 4);
  float* cF     = (float*)alloc((size_t)NB * ENC_H * 4);
  float* hR     = (float*)alloc((size_t)NB * ENC_H * 4);
  float* cR     = (float*)alloc((size_t)NB * ENC_H * 4);
  unsigned short* hencF = (unsigned short*)alloc((size_t)NB * ENC_H * 2);
  unsigned short* hencR = (unsigned short*)alloc((size_t)NB * ENC_H * 2);
  float* gatesF = (float*)alloc((size_t)NB * 2048 * 4);
  float* gatesR = (float*)alloc((size_t)NB * 2048 * 4);
  float* gatesD = (float*)alloc((size_t)NB * 4096 * 4);
  float* hdec   = (float*)alloc((size_t)NB * DEC_H * 4);
  float* cdec   = (float*)alloc((size_t)NB * DEC_H * 4);
  unsigned short* hq   = (unsigned short*)alloc((size_t)NB * DEC_H * 2);
  unsigned short* actb = (unsigned short*)alloc((size_t)NB * ACT_K * 2);
  unsigned short* catb = (unsigned short*)alloc((size_t)NB * 2048 * 2);
  float* qbuf   = (float*)alloc((size_t)NB * DEC_H * 4);
  float* logits = (float*)alloc((size_t)NB * GEN_NP * 4);

  auto conv = [&](const float* s, int R, int C, unsigned short* d, int ldd, int col0,
                  int Rp, int Cp) {
    long tot = (long)Rp * Cp;
    nmt_convert<<<(unsigned)((tot + 255) / 256), 256, 0, stream>>>(s, R, C, d, ldd, col0, Rp, Cp);
  };
  auto zero = [&](float* p, long n) {
    nmt_zero<<<(unsigned)((n + 255) / 256), 256, 0, stream>>>(p, n);
  };
#define GEMM(A, lda, Bm, ldb, b0, b1, nb, add, ldadd, of, ob, ldo, M, N, K, act)               \
  nmt_gemm_bf16<<<dim3((N) / TN, (M) / TM), 256, 0, stream>>>(                                 \
      A, lda, Bm, ldb, b0, b1, nb, add, ldadd, of, ob, ldo, K, act)

  // ---- weight conversion (f32 -> bf16, padded) ----
  conv(eWihF, 2048, EMB,  WihFb, EMB_P, 0, 2048, EMB_P);
  conv(eWihR, 2048, EMB,  WihRb, EMB_P, 0, 2048, EMB_P);
  conv(eWhhF, 2048, ENC_H, WhhFb, ENC_H, 0, 2048, ENC_H);
  conv(eWhhR, 2048, ENC_H, WhhRb, ENC_H, 0, 2048, ENC_H);
  conv(dWih, 4096, 1324, Wcomb, ACT_K, 0,      4096, INP_KP);
  conv(dWhh, 4096, 1024, Wcomb, ACT_K, INP_KP, 4096, 1024);
  conv(wi, 1024, 1024, wib, 1024, 0, 1024, 1024);
  conv(wo, 1024, 2048, wob, 2048, 0, 1024, 2048);
  conv(gen_w, TGT_V, 1024, genwb, 1024, 0, GEN_NP, 1024);

  // ---- src embedding gather -> bf16 (3200 x 320) ----
  nmt_embed_src<<<(unsigned)(((long)TSEQ * NB * EMB_P + 255) / 256), 256, 0, stream>>>(src, enc_emb, xbf);

  // ---- init states, zero output slice 0 ----
  zero(hF, NB * ENC_H); zero(cF, NB * ENC_H); zero(hR, NB * ENC_H); zero(cR, NB * ENC_H);
  zero((float*)hencF, NB * ENC_H / 2); zero((float*)hencR, NB * ENC_H / 2);
  zero(out, (long)NB * TGT_V);

  // ---- encoder input projections: xp = x @ Wih^T + bih + bhh ----
  GEMM(xbf, EMB_P, WihFb, EMB_P, ebihF, ebhhF, 2048, nullptr, 0, xpF, nullptr, 2048,
       TSEQ * NB, 2048, EMB_P, 0);
  GEMM(xbf, EMB_P, WihRb, EMB_P, ebihR, ebhhR, 2048, nullptr, 0, xpR, nullptr, 2048,
       TSEQ * NB, 2048, EMB_P, 0);

  // ---- encoder recurrence (both directions) ----
  for (int t = 0; t < TSEQ; ++t) {
    GEMM(hencF, ENC_H, WhhFb, ENC_H, nullptr, nullptr, 0, xpF + (long)t * NB * 2048, 2048,
         gatesF, nullptr, 2048, NB, 2048, ENC_H, 0);
    GEMM(hencR, ENC_H, WhhRb, ENC_H, nullptr, nullptr, 0, xpR + (long)(TSEQ - 1 - t) * NB * 2048, 2048,
         gatesR, nullptr, 2048, NB, 2048, ENC_H, 0);
    nmt_lstm_cell<<<NB * ENC_H / 256, 256, 0, stream>>>(gatesF, ENC_H, hF, cF,
        hencF, ENC_H, nullptr, 0, nullptr, 0, hsF + (long)t * NB * ENC_H);
    nmt_lstm_cell<<<NB * ENC_H / 256, 256, 0, stream>>>(gatesR, ENC_H, hR, cR,
        hencR, ENC_H, nullptr, 0, nullptr, 0, hsR + (long)t * NB * ENC_H);
  }

  // ---- enc_out concat + decoder h0/c0 interleave ----
  nmt_encout<<<(unsigned)(((long)TSEQ * NB * ENC_H + 255) / 256), 256, 0, stream>>>(hsF, hsR, eo);
  nmt_h0<<<NB * ENC_H / 256, 256, 0, stream>>>(hF, cF, hR, cR, hdec, cdec, hq, actb, catb);

  // ---- decoder: 49 steps ----
  for (int s = 0; s < TSEQ - 1; ++s) {
    // q = h @ wi^T
    GEMM(hq, DEC_H, wib, 1024, nullptr, nullptr, 0, nullptr, 0, qbuf, nullptr, DEC_H,
         NB, DEC_H, 1024, 0);
    // attention + context -> cat[:,0:1024]; pe -> act[:,1024:1344]
    nmt_attn<<<NB, 256, 0, stream>>>(qbuf, eo, dec_emb, trg + s * NB, catb, actb);
    // ct = tanh([st,h] @ wo^T) -> act[:,0:1024] (bf16)
    GEMM(catb, 2048, wob, 2048, nullptr, nullptr, 0, nullptr, 0, nullptr, actb, ACT_K,
         NB, 1024, 2048, 1);
    // gates = [inp,h] @ [Wih|Whh]^T + bih + bhh
    GEMM(actb, ACT_K, Wcomb, ACT_K, dbih, dbhh, 4096, nullptr, 0, gatesD, nullptr, 4096,
         NB, 4096, ACT_K, 0);
    // LSTM cell; scatter bf16 h to hq, act h-region, cat h-region
    nmt_lstm_cell<<<NB * DEC_H / 256, 256, 0, stream>>>(gatesD, DEC_H, hdec, cdec,
        hq, DEC_H, actb + INP_KP, ACT_K, catb + DEC_H, 2048, nullptr);
    // logits = h @ gen_w^T + gen_b
    GEMM(hq, DEC_H, genwb, 1024, gen_b, nullptr, TGT_V, nullptr, 0, logits, nullptr, GEN_NP,
         NB, GEN_NP, 1024, 0);
    // log_softmax -> out[s+1]
    nmt_logsoftmax<<<NB, 256, 0, stream>>>(logits, GEN_NP, TGT_V,
                                           out + (long)(s + 1) * NB * TGT_V, TGT_V);
  }
#undef GEMM
}
